// SPARCCrossCoder_47459388621169
// MI455X (gfx1250) — compile-verified, matrix operand order checked by reading, not run on hardware
//
#include <hip/hip_runtime.h>
#include <hip/hip_bf16.h>
#include <stdint.h>

// ---------------------------------------------------------------------------
// SPARC CrossCoder forward for MI455X (gfx1250, wave32, WMMA).
// Outputs (flat, f32): z_u, z_c, mask, x_u_hat, x_c_hat, x_u_cross, x_c_cross
// ---------------------------------------------------------------------------

typedef __attribute__((ext_vector_type(16))) __bf16 v16bf;
typedef __attribute__((ext_vector_type(8)))  float  v8f;

#define BATCH   4096
#define IN_DIM  1024
#define DICT    16384
#define KHALF   32
#define MAX_NNZ 128

#define TM 128
#define TN 128
#define TK 32
#define LSTR 20   // uints per LDS tile row: 16 data (32 bf16) + 4 pad -> 80B stride

// ---- helpers --------------------------------------------------------------

__device__ __forceinline__ uint32_t bf16_rne(float f) {
  uint32_t u = __float_as_uint(f);
  uint32_t r = u + 0x7FFFu + ((u >> 16) & 1u);   // round to nearest even
  return r >> 16;
}
__device__ __forceinline__ uint32_t pack_bf16(float lo, float hi) {
  return bf16_rne(lo) | (bf16_rne(hi) << 16);
}
// float -> order-preserving unsigned key (larger float => larger key)
__device__ __forceinline__ uint32_t f2key(float f) {
  uint32_t b = __float_as_uint(f);
  return (b & 0x80000000u) ? ~b : (b | 0x80000000u);
}
__device__ __forceinline__ v16bf load_frag(const uint32_t* rowp, int c) {
  union { uint32_t u[8]; v16bf v; } t;
  const uint4 lo = *reinterpret_cast<const uint4*>(rowp + c);       // K pairs c..c+3
  const uint4 hi = *reinterpret_cast<const uint4*>(rowp + c + 8);   // K pairs c+8..c+11
  t.u[0] = lo.x; t.u[1] = lo.y; t.u[2] = lo.z; t.u[3] = lo.w;
  t.u[4] = hi.x; t.u[5] = hi.y; t.u[6] = hi.z; t.u[7] = hi.w;
  return t.v;
}

// ---- 1) encoder GEMM:  H = X @ W + b   (M=4096, K=1024, N=16384) ----------
// 256 threads = 8 waves. WG tile 128x128. Wave (wr,wc) owns M:[wr*64,+64) x
// N:[wc*32,+32) => 4x2 WMMA 16x16x32 bf16 tiles accumulated in f32.

__global__ __launch_bounds__(256)
void encoder_gemm_kernel(const float* __restrict__ X,
                         const float* __restrict__ W,
                         const float* __restrict__ bias,
                         float* __restrict__ H) {
  __shared__ uint32_t aTile[TM * LSTR];   // 10240 B, packed bf16 pairs, K-major
  __shared__ uint32_t bTile[TN * LSTR];   // 10240 B, N-major rows of K

  const int tid    = threadIdx.x;
  const int lane   = tid & 31;
  const int wave   = tid >> 5;
  const int wr     = wave >> 2;        // 0..1
  const int wc     = wave & 3;         // 0..3
  const int laneLo = lane & 15;
  const int laneHi = lane >> 4;        // 0/1
  const int cOff   = laneHi * 4;       // fragment K-pair chunk base per ISA layout

  const int mBase = blockIdx.y * TM;
  const int nBase = blockIdx.x * TN;

  v8f acc[4][2] = {};

  for (int k0 = 0; k0 < IN_DIM; k0 += TK) {
    __syncthreads();
    // A tile: 128 rows x 32 K (f32) -> bf16 pairs. f = tid + i*256 in [0,1024)
#pragma unroll
    for (int i = 0; i < 4; ++i) {
      int f   = tid + i * 256;
      int row = f >> 3;                 // 8 float4 per row
      int seg = f & 7;
      const float4 v = *reinterpret_cast<const float4*>(
          X + (size_t)(mBase + row) * IN_DIM + k0 + seg * 4);
      uint2 p;
      p.x = pack_bf16(v.x, v.y);
      p.y = pack_bf16(v.z, v.w);
      *reinterpret_cast<uint2*>(&aTile[row * LSTR + seg * 2]) = p;   // b64 store
    }
    // B tile: 32 K-rows x 128 N, stored N-major. thread: one n, 4 consecutive k.
#pragma unroll
    for (int i = 0; i < 4; ++i) {
      int f  = tid + i * 256;
      int n  = f & 127;
      int kg = f >> 7;                  // 0..7 -> k = kg*4..kg*4+3
      const float* wp = W + (size_t)(k0 + kg * 4) * DICT + nBase + n;
      float w0 = wp[0];
      float w1 = wp[DICT];
      float w2 = wp[2 * DICT];
      float w3 = wp[3 * DICT];
      uint2 p;
      p.x = pack_bf16(w0, w1);
      p.y = pack_bf16(w2, w3);
      *reinterpret_cast<uint2*>(&bTile[n * LSTR + kg * 2]) = p;      // b64 store
    }
    __syncthreads();

    // B fragments (2 per wave), then 4 A fragments x 2 WMMAs each.
    v16bf bf0 = load_frag(&bTile[(wc * 32 + 0  + laneLo) * LSTR], cOff);
    v16bf bf1 = load_frag(&bTile[(wc * 32 + 16 + laneLo) * LSTR], cOff);
#pragma unroll
    for (int i = 0; i < 4; ++i) {
      v16bf af = load_frag(&aTile[(wr * 64 + i * 16 + laneLo) * LSTR], cOff);
      acc[i][0] = __builtin_amdgcn_wmma_f32_16x16x32_bf16(
          false, af, false, bf0, (short)0, acc[i][0], false, false);
      acc[i][1] = __builtin_amdgcn_wmma_f32_16x16x32_bf16(
          false, af, false, bf1, (short)0, acc[i][1], false, false);
    }
  }

  // epilogue: C/D layout -> VGPR r holds M = r (lanes 0-15) / 8+r (lanes 16-31)
#pragma unroll
  for (int i = 0; i < 4; ++i) {
#pragma unroll
    for (int j = 0; j < 2; ++j) {
      const int col = nBase + wc * 32 + j * 16 + laneLo;
      const float b = bias[col];
#pragma unroll
      for (int r = 0; r < 8; ++r) {
        const int row = mBase + wr * 64 + i * 16 + laneHi * 8 + r;
        H[(size_t)row * DICT + col] = acc[i][j][r] + b;
      }
    }
  }
}

// ---- 2) exact top-32 threshold per (row, stream) via radix select ---------

__global__ __launch_bounds__(256)
void topk_threshold_kernel(const float* __restrict__ Hu,
                           const float* __restrict__ Hc,
                           uint32_t* __restrict__ thr) {
  const int row = blockIdx.x;
  const int s   = blockIdx.y;
  const float* h = (s == 0 ? Hu : Hc) + (size_t)row * DICT;

  __shared__ unsigned int hist[256];
  __shared__ uint32_t cand[1024];
  __shared__ unsigned int candCount;
  __shared__ uint32_t prefS, kkS;

  const int tid = threadIdx.x;
  hist[tid] = 0;
  if (tid == 0) candCount = 0;
  __syncthreads();

  for (int i = tid; i < DICT; i += 256)
    atomicAdd(&hist[f2key(h[i]) >> 24], 1u);
  __syncthreads();

  if (tid == 0) {
    uint32_t cum = 0; int b = 255;
    for (; b > 0; --b) { uint32_t c = hist[b]; if (cum + c >= KHALF) break; cum += c; }
    prefS = (uint32_t)b << 24;
    kkS   = KHALF - cum;
  }
  __syncthreads();

  const uint32_t topByte = prefS >> 24;
  for (int i = tid; i < DICT; i += 256) {
    uint32_t key = f2key(h[i]);
    if ((key >> 24) == topByte) {
      uint32_t p = atomicAdd(&candCount, 1u);
      if (p < 1024u) cand[p] = key;
    }
  }
  __syncthreads();
  const uint32_t nc = candCount < 1024u ? candCount : 1024u;

  for (int shift = 20; shift >= 0; shift -= 4) {
    if (tid < 16) hist[tid] = 0;
    __syncthreads();
    const uint32_t pmask = ~((1u << (shift + 4)) - 1u);
    const uint32_t pref  = prefS;
    for (uint32_t i = tid; i < nc; i += 256) {
      uint32_t key = cand[i];
      if ((key & pmask) == pref) atomicAdd(&hist[(key >> shift) & 15u], 1u);
    }
    __syncthreads();
    if (tid == 0) {
      uint32_t kk = kkS, cum = 0; int d = 15;
      for (; d > 0; --d) { uint32_t c = hist[d]; if (cum + c >= kk) break; cum += c; }
      prefS |= (uint32_t)d << shift;
      kkS    = kk - cum;
    }
    __syncthreads();
  }
  if (tid == 0) thr[(size_t)s * BATCH + row] = prefS;   // key of k-th largest
}

// ---- 3) mask + relu*mask in place + deterministic ordered compaction ------

__global__ __launch_bounds__(256)
void mask_compact_kernel(float* __restrict__ Hu, float* __restrict__ Hc,
                         const uint32_t* __restrict__ thr,
                         float* __restrict__ maskOut,
                         int* __restrict__ nnz, int* __restrict__ idxL,
                         float* __restrict__ zuL, float* __restrict__ zcL) {
  const int row = blockIdx.x;
  const int tid = threadIdx.x;
  const uint32_t tu = thr[row];
  const uint32_t tc = thr[BATCH + row];
  float* hu = Hu + (size_t)row * DICT;
  float* hc = Hc + (size_t)row * DICT;
  float* mk = maskOut + (size_t)row * DICT;

  const int CH   = DICT / 256;       // 64 contiguous elements per thread
  const int base = tid * CH;
  unsigned long long bits = 0ull;
  int cnt = 0;

  for (int j = 0; j < CH; ++j) {
    const int i = base + j;
    const float vu = hu[i], vc = hc[i];
    const bool sel = (f2key(vu) >= tu) || (f2key(vc) >= tc);
    hu[i] = (sel && vu > 0.f) ? vu : 0.f;    // z_u = relu(h_u)*mask
    hc[i] = (sel && vc > 0.f) ? vc : 0.f;    // z_c = relu(h_c)*mask
    mk[i] = sel ? 1.f : 0.f;
    if (sel) { bits |= 1ull << j; ++cnt; }
  }

  __shared__ int cnts[256];
  __shared__ int offs[256];
  cnts[tid] = cnt;
  __syncthreads();
  if (tid == 0) {
    int acc = 0;
    for (int t = 0; t < 256; ++t) { offs[t] = acc; acc += cnts[t]; }
    nnz[row] = acc < MAX_NNZ ? acc : MAX_NNZ;
  }
  __syncthreads();

  int p = offs[tid];
  for (int j = 0; j < CH; ++j) {
    if (bits & (1ull << j)) {
      if (p < MAX_NNZ) {
        const int i = base + j;
        idxL[(size_t)row * MAX_NNZ + p] = i;
        zuL [(size_t)row * MAX_NNZ + p] = hu[i];
        zcL [(size_t)row * MAX_NNZ + p] = hc[i];
      }
      ++p;
    }
  }
}

// ---- 4) sparse decode: 4 outputs, shared W_dec gathers --------------------

__global__ __launch_bounds__(256)
void sparse_decode_kernel(const int* __restrict__ nnz, const int* __restrict__ idxL,
                          const float* __restrict__ zuL, const float* __restrict__ zcL,
                          const float* __restrict__ Wdu, const float* __restrict__ Wdc,
                          float* __restrict__ Xuh, float* __restrict__ Xch,
                          float* __restrict__ Xuc, float* __restrict__ Xcc) {
  const int row = blockIdx.x;
  const int tid = threadIdx.x;
  int n = nnz[row]; n = n > MAX_NNZ ? MAX_NNZ : n;

  __shared__ int   sI[MAX_NNZ];
  __shared__ float sU[MAX_NNZ];
  __shared__ float sC[MAX_NNZ];
  for (int i = tid; i < n; i += 256) {
    sI[i] = idxL[(size_t)row * MAX_NNZ + i];
    sU[i] = zuL [(size_t)row * MAX_NNZ + i];
    sC[i] = zcL [(size_t)row * MAX_NNZ + i];
  }
  __syncthreads();

  const int col = tid * 4;
  float uh0=0,uh1=0,uh2=0,uh3=0;   // x_u_hat   = z_u @ Wdu
  float ch0=0,ch1=0,ch2=0,ch3=0;   // x_c_hat   = z_c @ Wdc
  float uc0=0,uc1=0,uc2=0,uc3=0;   // x_u_cross = z_c @ Wdu
  float cc0=0,cc1=0,cc2=0,cc3=0;   // x_c_cross = z_u @ Wdc

  for (int e = 0; e < n; ++e) {
    const float zu = sU[e], zc = sC[e];
    const size_t wo = (size_t)sI[e] * IN_DIM + col;
    const float4 u4 = *reinterpret_cast<const float4*>(Wdu + wo);
    const float4 c4 = *reinterpret_cast<const float4*>(Wdc + wo);
    uh0 = fmaf(zu, u4.x, uh0); uh1 = fmaf(zu, u4.y, uh1);
    uh2 = fmaf(zu, u4.z, uh2); uh3 = fmaf(zu, u4.w, uh3);
    uc0 = fmaf(zc, u4.x, uc0); uc1 = fmaf(zc, u4.y, uc1);
    uc2 = fmaf(zc, u4.z, uc2); uc3 = fmaf(zc, u4.w, uc3);
    ch0 = fmaf(zc, c4.x, ch0); ch1 = fmaf(zc, c4.y, ch1);
    ch2 = fmaf(zc, c4.z, ch2); ch3 = fmaf(zc, c4.w, ch3);
    cc0 = fmaf(zu, c4.x, cc0); cc1 = fmaf(zu, c4.y, cc1);
    cc2 = fmaf(zu, c4.z, cc2); cc3 = fmaf(zu, c4.w, cc3);
  }

  const size_t o = (size_t)row * IN_DIM + col;
  *reinterpret_cast<float4*>(Xuh + o) = make_float4(uh0, uh1, uh2, uh3);
  *reinterpret_cast<float4*>(Xch + o) = make_float4(ch0, ch1, ch2, ch3);
  *reinterpret_cast<float4*>(Xuc + o) = make_float4(uc0, uc1, uc2, uc3);
  *reinterpret_cast<float4*>(Xcc + o) = make_float4(cc0, cc1, cc2, cc3);
}

// ---- launch ---------------------------------------------------------------

extern "C" void kernel_launch(void* const* d_in, const int* in_sizes, int n_in,
                              void* d_out, int out_size, void* d_ws, size_t ws_size,
                              hipStream_t stream) {
  (void)in_sizes; (void)n_in; (void)out_size; (void)ws_size;

  const float* x_u     = (const float*)d_in[0];
  const float* x_c     = (const float*)d_in[1];
  const float* W_enc_u = (const float*)d_in[2];
  const float* b_enc_u = (const float*)d_in[3];
  const float* W_enc_c = (const float*)d_in[4];
  const float* b_enc_c = (const float*)d_in[5];
  const float* W_dec_u = (const float*)d_in[6];
  const float* W_dec_c = (const float*)d_in[7];

  float* out = (float*)d_out;
  const size_t BD  = (size_t)BATCH * DICT;
  const size_t BIN = (size_t)BATCH * IN_DIM;
  float* Zu  = out;                 // h_u then z_u in place
  float* Zc  = out + BD;            // h_c then z_c in place
  float* Mk  = out + 2 * BD;
  float* Xuh = out + 3 * BD;
  float* Xch = Xuh + BIN;
  float* Xuc = Xch + BIN;
  float* Xcc = Xuc + BIN;

  uint8_t* ws = (uint8_t*)d_ws;
  uint32_t* thr  = (uint32_t*)ws;                                  //  32 KB
  int*      nnz  = (int*)(ws + 32768);                             //  16 KB
  int*      idxL = (int*)(ws + 32768 + 16384);                     //   2 MB
  float*    zuL  = (float*)(ws + 32768 + 16384 + 2097152);         //   2 MB
  float*    zcL  = (float*)(ws + 32768 + 16384 + 2 * 2097152);     //   2 MB

  dim3 gridE(DICT / TN, BATCH / TM);   // (128, 32)
  encoder_gemm_kernel<<<gridE, 256, 0, stream>>>(x_u, W_enc_u, b_enc_u, Zu);
  encoder_gemm_kernel<<<gridE, 256, 0, stream>>>(x_c, W_enc_c, b_enc_c, Zc);

  topk_threshold_kernel<<<dim3(BATCH, 2), 256, 0, stream>>>(Zu, Zc, thr);

  mask_compact_kernel<<<BATCH, 256, 0, stream>>>(Zu, Zc, thr, Mk,
                                                 nnz, idxL, zuL, zcL);

  sparse_decode_kernel<<<BATCH, 256, 0, stream>>>(nnz, idxL, zuL, zcL,
                                                  W_dec_u, W_dec_c,
                                                  Xuh, Xch, Xuc, Xcc);
}
